// ReprogrammingFuntion_70557722739002
// MI455X (gfx1250) — matrix operand order repaired and checked by reading, not compiled, not run on hardware
//
#include <hip/hip_runtime.h>
#include <hip/hip_bf16.h>
#include <math.h>

typedef __attribute__((ext_vector_type(16))) _Float16 v16h;
typedef __attribute__((ext_vector_type(8)))  _Float16 v8h;
typedef __attribute__((ext_vector_type(8)))  float    v8f;

#define B_      8
#define CIN     512
#define SEQ     40
#define EMB     300
#define EMBP    320               // EMB padded to multiple of 32
#define NFILT   256
#define IMG     384
#define NPR     24
#define NPTOT   576
#define EROWS   52                // LDS rows incl. zero guard (max row read = 50)
#define NSENT   (B_ * CIN)        // 4096
#define POOL    (3 * NFILT)       // 768
#define OUTIMG  (B_ * 3 * IMG * IMG)  // 3538944

// packed (padded, f16) weight segment sizes in halves
#define W2N (NFILT * 2 * EMBP)    // 163840
#define W3N (NFILT * 3 * EMBP)    // 245760
#define W4N (NFILT * 4 * EMBP)    // 327680
#define WTOT (W2N + W3N + W4N)    // 737280
#define NORM_OFF_BYTES ((size_t)WTOT * 2)  // 1474560

// ---------------------------------------------------------------------------
// Kernel 1: f32 conv weights -> f16, EMB padded 300->320 with zeros.
// Layout: Wk[f][q], q = j*320 + e, q < 320*k, row-major per filter.
// Also zeroes the norm accumulator.
// ---------------------------------------------------------------------------
__global__ void pack_weights(const float* __restrict__ w2,
                             const float* __restrict__ w3,
                             const float* __restrict__ w4,
                             _Float16* __restrict__ wp,
                             float* __restrict__ normAcc) {
    int g = blockIdx.x * blockDim.x + threadIdx.x;
    if (g == 0) *normAcc = 0.f;
    if (g >= WTOT) return;
    int kk, r;
    if (g < W2N)            { kk = 0; r = g; }
    else if (g < W2N + W3N) { kk = 1; r = g - W2N; }
    else                    { kk = 2; r = g - W2N - W3N; }
    const int k  = kk + 2;
    const int rl = EMBP * k;
    const int f  = r / rl;
    const int q  = r - f * rl;
    const int j  = q / EMBP;
    const int e  = q - j * EMBP;
    float v = 0.f;
    if (e < EMB) {
        const float* w = (kk == 0) ? w2 : (kk == 1) ? w3 : w4;
        v = w[(f * k + j) * EMB + e];
    }
    wp[g] = (_Float16)v;
}

// ---------------------------------------------------------------------------
// Tile worker, specialized on conv width K so the GEMM-K trip count
// (20/30/40 steps of 32) and segment offsets are compile-time scalars.
// ---------------------------------------------------------------------------
template <int K>
__device__ __forceinline__ void conv_tile(const _Float16* __restrict__ wpack,
                                          const _Float16* __restrict__ Epad,
                                          float* __restrict__ pool,
                                          const float* __restrict__ bias,
                                          int m, int nt, int lhi, int lmod) {
    constexpr int    ROWLEN = EMBP * K;          // GEMM K: 640 / 960 / 1280
    constexpr int    S      = ROWLEN / 32;       // 20 / 30 / 40
    constexpr size_t WOFF   = (K == 2) ? 0 : (K == 3) ? (size_t)W2N
                                                      : (size_t)(W2N + W3N);
    constexpr int    NK     = SEQ - K + 1;       // 39 / 38 / 37 valid columns
    constexpr int    POOLOFF = (K - 2) * NFILT;

    const int f = m * 16 + lmod;                 // this lane's A row (filter)
    const _Float16* __restrict__ Arow = wpack + WOFF + (size_t)f * ROWLEN;
    const int t = nt * 16 + lmod;                // this lane's B column (time)
    const _Float16* Bcol = &Epad[t * EMBP];

    v8f acc = {};
    #pragma unroll 2
    for (int s = 0; s < S; ++s) {
        const int q = s << 5;
        // A 16x32 f16 fragment: lanes<16 hold K {q..q+15};
        // lanes>=16 hold K {q+16..q+31}
        v8h alo = *(const v8h*)(Arow + q + 16 * lhi);
        v8h ahi = *(const v8h*)(Arow + q + 8 + 16 * lhi);
        v16h a = __builtin_shufflevector(alo, ahi,
                    0, 1, 2, 3, 4, 5, 6, 7, 8, 9, 10, 11, 12, 13, 14, 15);
        // B 32x16 f16 fragment: lane holds 16 contiguous K values of its
        // column; lanes<16 -> K {q..q+15}, lanes>=16 -> K {q+16..q+31}
        v16h b = *(const v16h*)(Bcol + q + 16 * lhi);
        acc = __builtin_amdgcn_wmma_f32_16x16x32_f16(
                  false, a, false, b, (short)0, acc, false, false);
    }

    if (t < NK) {
        #pragma unroll
        for (int r = 0; r < 8; ++r) {
            // C layout: lane holds M = r + 8*lhi, N = lmod
            const int fr = m * 16 + r + 8 * lhi;
            float v = acc[r] + bias[fr];
            v = fmaxf(v, 0.f);
            // bit-order == float order for non-negative IEEE floats
            atomicMax((int*)&pool[POOLOFF + fr], __float_as_int(v));
        }
    }
}

// ---------------------------------------------------------------------------
// Kernel 2: one workgroup (8 wave32) per sentence.
//  - gather 40 embedding rows into LDS (f16, 320-wide, zero guard rows)
//  - 3 GEMMs via v_wmma_f32_16x16x32_f16 (A = weights, B = im2col columns)
//  - bias + relu + max-over-time via LDS atomicMax
//  - scatter clipped patch into image, accumulate weighted squared norm
// ---------------------------------------------------------------------------
__global__ __launch_bounds__(256)
void textcnn_wmma(const int*      __restrict__ sent,
                  const float*    __restrict__ Wemb,
                  const _Float16* __restrict__ wpack,
                  const float*    __restrict__ b2,
                  const float*    __restrict__ b3,
                  const float*    __restrict__ b4,
                  float*          __restrict__ out,
                  float*          __restrict__ normAcc) {
    __shared__ __align__(32) _Float16 Epad[EROWS * EMBP];
    __shared__ float pool[POOL];
    __shared__ float blocksum;

    const int n    = blockIdx.x;
    const int tid  = threadIdx.x;
    const int bidx = n >> 9;       // batch index
    const int cidx = n & 511;      // channel / patch index

    const int* ids = sent + n * SEQ;

    // --- gather embeddings -> LDS (f16), pad cols 300..319 and rows 40..51 ---
    for (int i = tid; i < EROWS * EMBP; i += 256) {
        const int row = i / EMBP;
        const int e   = i - row * EMBP;
        float v = 0.f;
        if (row < SEQ && e < EMB) v = Wemb[(size_t)ids[row] * EMB + e];
        Epad[i] = (_Float16)v;
    }
    for (int i = tid; i < POOL; i += 256) pool[i] = 0.f;
    if (tid == 0) blocksum = 0.f;
    __syncthreads();

    // wave id is wave-uniform: pin it (and everything derived from it) to
    // SGPRs so the K loops get scalar bounds instead of EXEC-masked loops
    const int wave = __builtin_amdgcn_readfirstlane(tid >> 5);
    const int lane = tid & 31;
    const int lhi  = lane >> 4;    // half-wave: selects K sub-chunk
    const int lmod = lane & 15;    // row (A) / column (B) within tile

    // 144 tile jobs: 16 M-tiles x (3 k-sizes x 3 N-tiles)
    for (int job = wave; job < 144; job += 8) {
        const int m  = job / 9;
        const int c9 = job - m * 9;
        const int kk = c9 / 3;
        const int nt = c9 - kk * 3;
        if (kk == 0)      conv_tile<2>(wpack, Epad, pool, b2, m, nt, lhi, lmod);
        else if (kk == 1) conv_tile<3>(wpack, Epad, pool, b3, m, nt, lhi, lmod);
        else              conv_tile<4>(wpack, Epad, pool, b4, m, nt, lhi, lmod);
    }
    __syncthreads();

    // --- epilogue: norm contribution + clipped patch scatter ---
    const float wN = (cidx == 511) ? 65.f : 1.f;  // channel 511 tiles 65 patches
    float psum = 0.f;
    for (int i = tid; i < POOL; i += 256) {
        const float v  = pool[i];
        psum += wN * v * v;
        const float cl = fminf(v, 1.f);           // v >= 0 already (post-relu)
        const int ch = i >> 8;
        const int fe = i & 255;
        const int pi = fe >> 4;
        const int pj = fe & 15;
        {
            const int p   = cidx;
            const int row = (p / NPR) * 16 + pi;
            const int col = (p % NPR) * 16 + pj;
            out[(((size_t)bidx * 3 + ch) * IMG + row) * IMG + col] = cl;
        }
        if (cidx == 511) {
            for (int p = 512; p < NPTOT; ++p) {
                const int row = (p / NPR) * 16 + pi;
                const int col = (p % NPR) * 16 + pj;
                out[(((size_t)bidx * 3 + ch) * IMG + row) * IMG + col] = cl;
            }
        }
    }
    atomicAdd(&blocksum, psum);
    __syncthreads();
    if (tid == 0) atomicAdd(normAcc, blocksum);
}

// ---------------------------------------------------------------------------
// Kernel 3: scalar pert_norm = sqrt(sum of squares) / B
// ---------------------------------------------------------------------------
__global__ void finalize_norm(const float* __restrict__ normAcc,
                              float* __restrict__ out) {
    out[OUTIMG] = sqrtf(*normAcc) / (float)B_;
}

extern "C" void kernel_launch(void* const* d_in, const int* in_sizes, int n_in,
                              void* d_out, int out_size, void* d_ws, size_t ws_size,
                              hipStream_t stream) {
    (void)in_sizes; (void)n_in; (void)out_size; (void)ws_size;
    const int*   sent = (const int*)d_in[0];
    const float* Wemb = (const float*)d_in[1];
    const float* w2   = (const float*)d_in[2];
    const float* bi2  = (const float*)d_in[3];
    const float* w3   = (const float*)d_in[4];
    const float* bi3  = (const float*)d_in[5];
    const float* w4   = (const float*)d_in[6];
    const float* bi4  = (const float*)d_in[7];
    float*     out    = (float*)d_out;
    _Float16*  wpack  = (_Float16*)d_ws;
    float*     normAcc = (float*)((char*)d_ws + NORM_OFF_BYTES);

    pack_weights<<<(WTOT + 255) / 256, 256, 0, stream>>>(w2, w3, w4, wpack, normAcc);
    textcnn_wmma<<<NSENT, 256, 0, stream>>>(sent, Wemb, wpack, bi2, bi3, bi4,
                                            out, normAcc);
    finalize_norm<<<1, 1, 0, stream>>>(normAcc, out);
}
